// RWKV7TimeMix_14869176778903
// MI455X (gfx1250) — matrix operand order repaired and checked by reading, not compiled, not run on hardware
//
#include <hip/hip_runtime.h>
#include <hip/hip_bf16.h>
#include <math.h>

// ---- problem constants (match reference) ----
#define BB 2
#define TT 4096
#define CC 2048
#define HH 32
#define HN 64
#define MM (BB * TT)          // 8192 rows
#define LN_EPS 0.00064f

typedef __bf16 bf16_t;
typedef __attribute__((ext_vector_type(16))) __bf16 v16bf;
typedef __attribute__((ext_vector_type(8)))  __bf16 v8bf;
typedef __attribute__((ext_vector_type(8)))  float  v8f;

// --------------------------- async-to-LDS support ---------------------------
#if defined(__has_builtin)
#if __has_builtin(__builtin_amdgcn_global_load_async_to_lds_b32) && \
    __has_builtin(__builtin_amdgcn_s_wait_asynccnt)
#define HAVE_ASYNC_LDS 1
#endif
#endif

#ifdef HAVE_ASYNC_LDS
// builtin signature: (AS1 int* gaddr, AS3 int* ldsaddr, imm offset, imm cpol)
#define ASYNC_CP(gp, lp)                                                     \
  __builtin_amdgcn_global_load_async_to_lds_b32(                             \
      (__attribute__((address_space(1))) int*)(gp),                          \
      (__attribute__((address_space(3))) int*)(lp), 0, 0)
#define ASYNC_WAIT(n) __builtin_amdgcn_s_wait_asynccnt(n)
#else
#define ASYNC_CP(gp, lp) (*(lp) = *(gp))
#define ASYNC_WAIT(n)
#endif

// ---------------------------------------------------------------------------
// WMMA fragment helpers (CDNA5 16x16x32 bf16, wave32)
// ---------------------------------------------------------------------------
__device__ __forceinline__ v16bf load_a_frag(const bf16_t* __restrict__ A,
                                             int lda, int row0, int k0, int lane) {
  int m = lane & 15, half = lane >> 4;
  const bf16_t* p = A + (size_t)(row0 + m) * lda + k0 + half * 8;
  union { v16bf v; v8bf h[2]; } u;
  u.h[0] = *(const v8bf*)(p);       // K base..base+7
  u.h[1] = *(const v8bf*)(p + 16);  // K base+16..base+23
  return u.v;
}

__device__ __forceinline__ v16bf load_b_frag(const bf16_t* __restrict__ B,
                                             int ldb, int col0, int k0, int lane) {
  int n = lane & 15, half = lane >> 4;
  const bf16_t* p = B + (size_t)(col0 + n) * ldb + k0 + half * 8;
  union { v16bf v; v8bf h[2]; } u;
  u.h[0] = *(const v8bf*)(p);
  u.h[1] = *(const v8bf*)(p + 16);
  return u.v;
}

// ---------------------------------------------------------------------------
// GEMM: C[M,N](f32) = A[M,K](bf16,row-major) * B[N,K](bf16,row-major)^T
// block (32,4): each wave 64x32 (4x2 tiles); block tile 128x64.
// grid = (N/64, M/128). Requires K % 64 == 0. Double-buffered k-loop.
// ---------------------------------------------------------------------------
__global__ __launch_bounds__(128) void gemm_bf16_nt(
    const bf16_t* __restrict__ A, const bf16_t* __restrict__ B,
    float* __restrict__ C, int M, int N, int K) {
  const int lane = threadIdx.x;
  const int wave = threadIdx.y;
  const int wm = wave >> 1, wn = wave & 1;
  const int row0 = blockIdx.y * 128 + wm * 64;
  const int col0 = blockIdx.x * 64 + wn * 32;

  v8f acc[4][2] = {};
  v16bf afA[4], bfA[2], afB[4], bfB[2];

#pragma unroll
  for (int i = 0; i < 4; ++i) afA[i] = load_a_frag(A, K, row0 + 16 * i, 0, lane);
#pragma unroll
  for (int j = 0; j < 2; ++j) bfA[j] = load_b_frag(B, K, col0 + 16 * j, 0, lane);

  for (int k0 = 0; k0 < K; k0 += 64) {
    // prefetch second 32-chunk while first computes
#pragma unroll
    for (int i = 0; i < 4; ++i) afB[i] = load_a_frag(A, K, row0 + 16 * i, k0 + 32, lane);
#pragma unroll
    for (int j = 0; j < 2; ++j) bfB[j] = load_b_frag(B, K, col0 + 16 * j, k0 + 32, lane);
#pragma unroll
    for (int i = 0; i < 4; ++i)
#pragma unroll
      for (int j = 0; j < 2; ++j)
        acc[i][j] = __builtin_amdgcn_wmma_f32_16x16x32_bf16(
            false, afA[i], false, bfA[j], (short)0, acc[i][j], false, false);

    if (k0 + 64 < K) {
#pragma unroll
      for (int i = 0; i < 4; ++i) afA[i] = load_a_frag(A, K, row0 + 16 * i, k0 + 64, lane);
#pragma unroll
      for (int j = 0; j < 2; ++j) bfA[j] = load_b_frag(B, K, col0 + 16 * j, k0 + 64, lane);
    }
#pragma unroll
    for (int i = 0; i < 4; ++i)
#pragma unroll
      for (int j = 0; j < 2; ++j)
        acc[i][j] = __builtin_amdgcn_wmma_f32_16x16x32_bf16(
            false, afB[i], false, bfB[j], (short)0, acc[i][j], false, false);
  }

  const int n = lane & 15, half = lane >> 4;
#pragma unroll
  for (int i = 0; i < 4; ++i)
#pragma unroll
    for (int j = 0; j < 2; ++j) {
      float* p = C + (size_t)(row0 + 16 * i + half * 8) * N + col0 + 16 * j + n;
#pragma unroll
      for (int e = 0; e < 8; ++e) p[(size_t)e * N] = acc[i][j][e];
    }
}

// ---------------------------------------------------------------------------
// token shift + 6 mixes, f32 -> bf16
// ---------------------------------------------------------------------------
__global__ __launch_bounds__(256) void prep_kernel(
    const float* __restrict__ x,
    const float* __restrict__ cr, const float* __restrict__ cw,
    const float* __restrict__ ck, const float* __restrict__ cv,
    const float* __restrict__ ca, const float* __restrict__ cg,
    bf16_t* __restrict__ XR, bf16_t* __restrict__ XW, bf16_t* __restrict__ XK,
    bf16_t* __restrict__ XV, bf16_t* __restrict__ XA, bf16_t* __restrict__ XG) {
  size_t idx = (size_t)blockIdx.x * blockDim.x + threadIdx.x;
  if (idx >= (size_t)MM * CC) return;
  int c = (int)(idx % CC);
  size_t bt = idx / CC;
  int t = (int)(bt % TT);
  float xv = x[idx];
  float prev = (t == 0) ? 0.0f : x[idx - CC];
  float xx = prev - xv;
  XR[idx] = (bf16_t)(xv + xx * cr[c]);
  XW[idx] = (bf16_t)(xv + xx * cw[c]);
  XK[idx] = (bf16_t)(xv + xx * ck[c]);
  XV[idx] = (bf16_t)(xv + xx * cv[c]);
  XA[idx] = (bf16_t)(xv + xx * ca[c]);
  XG[idx] = (bf16_t)(xv + xx * cg[c]);
}

__global__ __launch_bounds__(256) void f32_to_bf16_kernel(
    const float* __restrict__ in, bf16_t* __restrict__ out, int n) {
  int i = blockIdx.x * blockDim.x + threadIdx.x;
  if (i < n) out[i] = (bf16_t)in[i];
}

// out[Npad,Kpad](bf16) = transpose(in[inRows,inCols]) zero padded
__global__ __launch_bounds__(256) void transpose_pad_kernel(
    const float* __restrict__ in, bf16_t* __restrict__ out,
    int inRows, int inCols, int Kpad, int Npad) {
  int i = blockIdx.x * blockDim.x + threadIdx.x;
  if (i >= Npad * Kpad) return;
  int nn = i / Kpad, k = i % Kpad;
  float v = (k < inRows && nn < inCols) ? in[(size_t)k * inCols + nn] : 0.0f;
  out[i] = (bf16_t)v;
}

// mode: 0 tanh, 1 identity, 2 sigmoid;   f32 -> bf16
__global__ __launch_bounds__(256) void act_kernel(
    const float* __restrict__ in, bf16_t* __restrict__ out, int n, int mode) {
  int i = blockIdx.x * blockDim.x + threadIdx.x;
  if (i >= n) return;
  float v = in[i];
  if (mode == 0) v = tanhf(v);
  else if (mode == 2) v = 1.0f / (1.0f + expf(-v));
  out[i] = (bf16_t)v;
}

// ---------------------------------------------------------------------------
// pre-wkv: decay, a, v-mix, k update, kk normalize; one wave per (token,head)
// ---------------------------------------------------------------------------
__global__ __launch_bounds__(256) void prewkv_kernel(
    float* __restrict__ k, float* __restrict__ v,
    float* __restrict__ wlin, float* __restrict__ alin,
    const float* __restrict__ vlin, const float* __restrict__ v_first,
    const float* __restrict__ w0, const float* __restrict__ a0,
    const float* __restrict__ v0, const float* __restrict__ k_k,
    const float* __restrict__ k_a,
    float* __restrict__ aa, float* __restrict__ bb) {
  int wave = threadIdx.x >> 5, lane = threadIdx.x & 31;
  long gid = (long)blockIdx.x * 8 + wave;   // bt*HH + h
  int h = (int)(gid % HH);
  size_t bt = (size_t)(gid / HH);
  size_t base = bt * CC + (size_t)h * HN;
  int ch0 = h * HN + lane, ch1 = ch0 + 32;
  size_t i0 = base + lane, i1 = i0 + 32;

  float k0v = k[i0], k1v = k[i1];
  float kk0 = k0v * k_k[ch0], kk1 = k1v * k_k[ch1];
  float ss = kk0 * kk0 + kk1 * kk1;
#pragma unroll
  for (int off = 16; off > 0; off >>= 1) ss += __shfl_xor(ss, off, 32);
  float inv = 1.0f / fmaxf(sqrtf(ss), 1e-12f);
  kk0 *= inv; kk1 *= inv;

  float av0 = 1.0f / (1.0f + expf(-(a0[ch0] + alin[i0])));
  float av1 = 1.0f / (1.0f + expf(-(a0[ch1] + alin[i1])));
  alin[i0] = av0; alin[i1] = av1;

  float z0 = w0[ch0] + wlin[i0], z1 = w0[ch1] + wlin[i1];
  float wv0 = -log1pf(expf(-z0)) - 0.5f;       // -softplus(-z) - 0.5
  float wv1 = -log1pf(expf(-z1)) - 0.5f;
  wlin[i0] = expf(-expf(wv0)); wlin[i1] = expf(-expf(wv1));

  float s0 = 1.0f / (1.0f + expf(-(v0[ch0] + vlin[i0])));
  float s1 = 1.0f / (1.0f + expf(-(v0[ch1] + vlin[i1])));
  float vv0 = v[i0], vv1 = v[i1];
  v[i0] = vv0 + (v_first[i0] - vv0) * s0;
  v[i1] = vv1 + (v_first[i1] - vv1) * s1;

  k[i0] = k0v * (1.0f + (av0 - 1.0f) * k_a[ch0]);
  k[i1] = k1v * (1.0f + (av1 - 1.0f) * k_a[ch1]);

  aa[i0] = -kk0; aa[i1] = -kk1;
  bb[i0] = kk0 * av0; bb[i1] = kk1 * av1;
}

// ---------------------------------------------------------------------------
// wkv7 scan: one block (64 threads) per (b,h); thread owns state row (64 f32).
// Per-step vectors staged into double-buffered LDS with async global->LDS
// copies (ASYNCcnt) so t+1's loads overlap t's math.
// ---------------------------------------------------------------------------
__global__ __launch_bounds__(64) void wkv7_scan_kernel(
    const float* __restrict__ r, const float* __restrict__ wd,
    const float* __restrict__ k, const float* __restrict__ v,
    const float* __restrict__ aa, const float* __restrict__ bb,
    float* __restrict__ y) {
  int bh = blockIdx.x;
  int b = bh / HH, h = bh % HH;
  int tid = threadIdx.x;                       // state row index / channel
  __shared__ float sbuf[2][6][HN];
  float S[HN];
#pragma unroll
  for (int i = 0; i < HN; ++i) S[i] = 0.0f;
  size_t base = (size_t)b * TT * CC + (size_t)h * HN;

  // stage t=0 into buffer 0
  {
    size_t idx = base + tid;
    ASYNC_CP(r + idx,  &sbuf[0][0][tid]);
    ASYNC_CP(wd + idx, &sbuf[0][1][tid]);
    ASYNC_CP(k + idx,  &sbuf[0][2][tid]);
    ASYNC_CP(v + idx,  &sbuf[0][3][tid]);
    ASYNC_CP(aa + idx, &sbuf[0][4][tid]);
    ASYNC_CP(bb + idx, &sbuf[0][5][tid]);
  }

  for (int t = 0; t < TT; ++t) {
    int cur = t & 1;
    size_t idx = base + (size_t)t * CC + tid;
    if (t + 1 < TT) {
      size_t nidx = idx + CC;
      int nxt = cur ^ 1;
      ASYNC_CP(r + nidx,  &sbuf[nxt][0][tid]);
      ASYNC_CP(wd + nidx, &sbuf[nxt][1][tid]);
      ASYNC_CP(k + nidx,  &sbuf[nxt][2][tid]);
      ASYNC_CP(v + nidx,  &sbuf[nxt][3][tid]);
      ASYNC_CP(aa + nidx, &sbuf[nxt][4][tid]);
      ASYNC_CP(bb + nidx, &sbuf[nxt][5][tid]);
      ASYNC_WAIT(6);  // the older 6 (current buffer) are complete
    } else {
      ASYNC_WAIT(0);
    }
    __syncthreads();

    const float* sr = sbuf[cur][0];
    const float* sw = sbuf[cur][1];
    const float* sk = sbuf[cur][2];
    const float* sv = sbuf[cur][3];
    const float* sa = sbuf[cur][4];
    const float* sb = sbuf[cur][5];

    float sacc = 0.0f;
#pragma unroll
    for (int j = 0; j < HN; ++j) sacc = fmaf(S[j], sa[j], sacc);
    float vv = sv[tid];
    float yv = 0.0f;
#pragma unroll
    for (int j = 0; j < HN; ++j) {
      float s = fmaf(S[j], sw[j], fmaf(sacc, sb[j], vv * sk[j]));
      S[j] = s;
      yv = fmaf(s, sr[j], yv);
    }
    y[idx] = yv;
    __syncthreads();  // everyone done with sbuf[cur] before it is re-filled
  }
}

// ---------------------------------------------------------------------------
// groupnorm(H) + bonus + gate -> bf16; one wave per (token,head)
// ---------------------------------------------------------------------------
__global__ __launch_bounds__(256) void post_kernel(
    const float* __restrict__ y, const float* __restrict__ r,
    const float* __restrict__ k, const float* __restrict__ v,
    const float* __restrict__ g, const float* __restrict__ r_k,
    const float* __restrict__ ln_w, const float* __restrict__ ln_b,
    bf16_t* __restrict__ yg) {
  int wave = threadIdx.x >> 5, lane = threadIdx.x & 31;
  long gid = (long)blockIdx.x * 8 + wave;
  int h = (int)(gid % HH);
  size_t bt = (size_t)(gid / HH);
  size_t base = bt * CC + (size_t)h * HN;
  int ch0 = h * HN + lane, ch1 = ch0 + 32;
  size_t i0 = base + lane, i1 = i0 + 32;

  float y0 = y[i0], y1 = y[i1];
  float s = y0 + y1, s2 = y0 * y0 + y1 * y1;
  float r0 = r[i0], r1 = r[i1], k0 = k[i0], k1 = k[i1];
  float d = r0 * k0 * r_k[ch0] + r1 * k1 * r_k[ch1];
#pragma unroll
  for (int off = 16; off > 0; off >>= 1) {
    s  += __shfl_xor(s,  off, 32);
    s2 += __shfl_xor(s2, off, 32);
    d  += __shfl_xor(d,  off, 32);
  }
  float mu  = s * (1.0f / HN);
  float var = s2 * (1.0f / HN) - mu * mu;
  float inv = 1.0f / sqrtf(var + LN_EPS);
  float o0 = (y0 - mu) * inv * ln_w[ch0] + ln_b[ch0] + d * v[i0];
  float o1 = (y1 - mu) * inv * ln_w[ch1] + ln_b[ch1] + d * v[i1];
  yg[i0] = (bf16_t)(o0 * g[i0]);
  yg[i1] = (bf16_t)(o1 * g[i1]);
}

// ---------------------------------------------------------------------------
extern "C" void kernel_launch(void* const* d_in, const int* in_sizes, int n_in,
                              void* d_out, int out_size, void* d_ws, size_t ws_size,
                              hipStream_t stream) {
  (void)in_sizes; (void)n_in; (void)out_size; (void)ws_size;
  const float* x   = (const float*)d_in[0];
  const float* v1f = (const float*)d_in[1];
  const float* x_r = (const float*)d_in[2];
  const float* x_w = (const float*)d_in[3];
  const float* x_k = (const float*)d_in[4];
  const float* x_v = (const float*)d_in[5];
  const float* x_a = (const float*)d_in[6];
  const float* x_g = (const float*)d_in[7];
  const float* w0  = (const float*)d_in[8];
  const float* w1  = (const float*)d_in[9];
  const float* w2  = (const float*)d_in[10];
  const float* a0  = (const float*)d_in[11];
  const float* a1  = (const float*)d_in[12];
  const float* a2  = (const float*)d_in[13];
  const float* v0  = (const float*)d_in[14];
  const float* v1  = (const float*)d_in[15];
  const float* v2  = (const float*)d_in[16];
  const float* g1  = (const float*)d_in[17];
  const float* g2  = (const float*)d_in[18];
  const float* k_k = (const float*)d_in[19];
  const float* k_a = (const float*)d_in[20];
  const float* r_k = (const float*)d_in[21];
  const float* W_r = (const float*)d_in[22];
  const float* W_k = (const float*)d_in[23];
  const float* W_v = (const float*)d_in[24];
  const float* W_o = (const float*)d_in[25];
  const float* lnw = (const float*)d_in[26];
  const float* lnb = (const float*)d_in[27];
  float* out = (float*)d_out;

  char* ws = (char*)d_ws;
  size_t off = 0;
  auto alloc = [&](size_t bytes) -> void* {
    void* p = ws + off;
    off += (bytes + 255) & ~(size_t)255;
    return p;
  };
  const size_t MC = (size_t)MM * CC;

  bf16_t* XR = (bf16_t*)alloc(MC * 2);
  bf16_t* XW = (bf16_t*)alloc(MC * 2);
  bf16_t* XK = (bf16_t*)alloc(MC * 2);
  bf16_t* XV = (bf16_t*)alloc(MC * 2);
  bf16_t* XA = (bf16_t*)alloc(MC * 2);
  bf16_t* XG = (bf16_t*)alloc(MC * 2);
  bf16_t* WRb = (bf16_t*)alloc((size_t)CC * CC * 2);
  bf16_t* WKb = (bf16_t*)alloc((size_t)CC * CC * 2);
  bf16_t* WVb = (bf16_t*)alloc((size_t)CC * CC * 2);
  bf16_t* WOb = (bf16_t*)alloc((size_t)CC * CC * 2);
  bf16_t* w1t = (bf16_t*)alloc((size_t)128 * CC * 2);   // [128, 2048]
  bf16_t* a1t = (bf16_t*)alloc((size_t)128 * CC * 2);
  bf16_t* v1t = (bf16_t*)alloc((size_t)64 * CC * 2);    // [64, 2048]
  bf16_t* g1t = (bf16_t*)alloc((size_t)256 * CC * 2);   // [256, 2048]
  bf16_t* w2t = (bf16_t*)alloc((size_t)CC * 128 * 2);   // [2048, 128]
  bf16_t* a2t = (bf16_t*)alloc((size_t)CC * 128 * 2);
  bf16_t* v2t = (bf16_t*)alloc((size_t)CC * 64 * 2);
  bf16_t* g2t = (bf16_t*)alloc((size_t)CC * 256 * 2);
  float* r_f = (float*)alloc(MC * 4);
  float* k_f = (float*)alloc(MC * 4);
  float* v_f = (float*)alloc(MC * 4);
  float* hw  = (float*)alloc((size_t)MM * 128 * 4);
  float* ha  = (float*)alloc((size_t)MM * 128 * 4);
  float* hv  = (float*)alloc((size_t)MM * 64 * 4);
  float* hg  = (float*)alloc((size_t)MM * 256 * 4);
  bf16_t* hwb = (bf16_t*)alloc((size_t)MM * 128 * 2);
  bf16_t* hab = (bf16_t*)alloc((size_t)MM * 128 * 2);
  bf16_t* hvb = (bf16_t*)alloc((size_t)MM * 64 * 2);
  bf16_t* hgb = (bf16_t*)alloc((size_t)MM * 256 * 2);
  float* wlin = (float*)alloc(MC * 4);
  float* alin = (float*)alloc(MC * 4);
  float* vlin = (float*)alloc(MC * 4);
  float* glin = (float*)alloc(MC * 4);
  float* aaB  = (float*)alloc(MC * 4);
  float* bbB  = (float*)alloc(MC * 4);
  float* yB   = (float*)alloc(MC * 4);
  bf16_t* ygb = (bf16_t*)alloc(MC * 2);

  // 1) token shift + mixes -> bf16
  prep_kernel<<<(unsigned)(MC / 256), 256, 0, stream>>>(
      x, x_r, x_w, x_k, x_v, x_a, x_g, XR, XW, XK, XV, XA, XG);

  // 2) weight conversion
  {
    int n = CC * CC, g = (n + 255) / 256;
    f32_to_bf16_kernel<<<g, 256, 0, stream>>>(W_r, WRb, n);
    f32_to_bf16_kernel<<<g, 256, 0, stream>>>(W_k, WKb, n);
    f32_to_bf16_kernel<<<g, 256, 0, stream>>>(W_v, WVb, n);
    f32_to_bf16_kernel<<<g, 256, 0, stream>>>(W_o, WOb, n);
  }
  auto tp = [&](const float* in, bf16_t* o, int inRows, int inCols, int Kpad, int Npad) {
    int n = Kpad * Npad;
    transpose_pad_kernel<<<(n + 255) / 256, 256, 0, stream>>>(in, o, inRows, inCols, Kpad, Npad);
  };
  tp(w1, w1t, CC, 96, CC, 128);   tp(w2, w2t, 96, CC, 128, CC);
  tp(a1, a1t, CC, 96, CC, 128);   tp(a2, a2t, 96, CC, 128, CC);
  tp(v1, v1t, CC, 64, CC, 64);    tp(v2, v2t, 64, CC, 64, CC);
  tp(g1, g1t, CC, 256, CC, 256);  tp(g2, g2t, 256, CC, 256, CC);

  // 3) first-stage GEMMs
  dim3 blk(32, 4);
  auto gemm = [&](const bf16_t* A, const bf16_t* Bm, float* Cm, int M, int N, int K) {
    dim3 grd(N / 64, M / 128);
    gemm_bf16_nt<<<grd, blk, 0, stream>>>(A, Bm, Cm, M, N, K);
  };
  gemm(XR, WRb, r_f, MM, CC, CC);
  gemm(XK, WKb, k_f, MM, CC, CC);
  gemm(XV, WVb, v_f, MM, CC, CC);
  gemm(XW, w1t, hw, MM, 128, CC);
  gemm(XA, a1t, ha, MM, 128, CC);
  gemm(XV, v1t, hv, MM, 64, CC);
  gemm(XG, g1t, hg, MM, 256, CC);

  // 4) hidden activations -> bf16
  act_kernel<<<(MM * 128 + 255) / 256, 256, 0, stream>>>(hw, hwb, MM * 128, 0); // tanh
  act_kernel<<<(MM * 128 + 255) / 256, 256, 0, stream>>>(ha, hab, MM * 128, 1); // id
  act_kernel<<<(MM * 64 + 255) / 256, 256, 0, stream>>>(hv, hvb, MM * 64, 1);   // id
  act_kernel<<<(MM * 256 + 255) / 256, 256, 0, stream>>>(hg, hgb, MM * 256, 2); // sigmoid

  // 5) second-stage GEMMs
  gemm(hwb, w2t, wlin, MM, CC, 128);
  gemm(hab, a2t, alin, MM, CC, 128);
  gemm(hvb, v2t, vlin, MM, CC, 64);
  gemm(hgb, g2t, glin, MM, CC, 256);

  // 6) pre-wkv elementwise + per-head kk normalization
  prewkv_kernel<<<(MM * HH) / 8, 256, 0, stream>>>(
      k_f, v_f, wlin, alin, vlin, v1f, w0, a0, v0, k_k, k_a, aaB, bbB);

  // 7) sequential delta-rule scan (async-LDS pipelined)
  wkv7_scan_kernel<<<BB * HH, HN, 0, stream>>>(r_f, wlin, k_f, v_f, aaB, bbB, yB);

  // 8) groupnorm + bonus + gate -> bf16
  post_kernel<<<(MM * HH) / 8, 256, 0, stream>>>(
      yB, r_f, k_f, v_f, glin, r_k, lnw, lnb, ygb);

  // 9) output projection
  gemm(ygb, WOb, out, MM, CC, CC);
}
